// SwinTransformerBlock_14645838479799
// MI455X (gfx1250) — matrix operand order repaired and checked by reading, not compiled, not run on hardware
//
#include <hip/hip_runtime.h>
#include <math.h>

typedef __attribute__((ext_vector_type(16))) _Float16 v16h;
typedef __attribute__((ext_vector_type(8)))  float    v8f;

// ---- static config (matches reference) ----
constexpr int Bn = 4, Tn = 6, Hn = 64, Wn = 64, Cn = 120;
constexpr int HEADS = 6, HD = 20;
constexpr int NW_PER_B = 192;          // windows per batch: 3*8*8
constexpr int NWIN = Bn * NW_PER_B;    // 768 windows
constexpr int NTOK = 128;              // tokens per window (2*8*8)
constexpr int MTOT = NWIN * NTOK;      // 98304 rows
constexpr int HID  = 240;

__device__ __forceinline__ v8f wmma_f16(v16h a, v16h b, v8f c) {
  return __builtin_amdgcn_wmma_f32_16x16x32_f16(false, a, false, b, (short)0, c, false, false);
}

// window-order row m  ->  token-major index (applies window reverse + unshift)
__device__ __forceinline__ size_t win_row_to_token(int m) {
  int widx = m >> 7, n = m & 127;
  int b   = widx / NW_PER_B, rem = widx % NW_PER_B;
  int tw  = rem >> 6, rem2 = rem & 63, hw = rem2 >> 3, ww = rem2 & 7;
  int wt  = n >> 6, wh = (n >> 3) & 7, wl = n & 7;
  int ts  = tw * 2 + wt, hs = hw * 8 + wh, zs = ww * 8 + wl;
  int to  = ts + 1; if (to >= Tn) to -= Tn;   // +shift (t)
  int ho  = (hs + 4) & 63;                    // +shift (h)
  int wo  = (zs + 4) & 63;                    // +shift (w)
  return ((size_t)((b * Tn + to) * Hn + ho)) * Wn + wo;
}

// ---- LayerNorm: one block (128 thr) per row; window_order selects shifted/window mapping ----
__global__ void ln_kernel(const float* __restrict__ xin,
                          const float* __restrict__ g, const float* __restrict__ bta,
                          _Float16* __restrict__ outh, int window_order) {
  __shared__ float red[128];
  int m = blockIdx.x, c = threadIdx.x;
  size_t tok = window_order ? win_row_to_token(m) : (size_t)m;
  float v = (c < Cn) ? xin[tok * Cn + c] : 0.f;
  red[c] = v; __syncthreads();
  for (int s = 64; s > 0; s >>= 1) { if (c < s) red[c] += red[c + s]; __syncthreads(); }
  float mean = red[0] * (1.f / Cn); __syncthreads();
  float d = (c < Cn) ? (v - mean) : 0.f;
  red[c] = d * d; __syncthreads();
  for (int s = 64; s > 0; s >>= 1) { if (c < s) red[c] += red[c + s]; __syncthreads(); }
  float rs = rsqrtf(red[0] * (1.f / Cn) + 1e-5f);
  if (c < Cn) outh[(size_t)m * Cn + c] = (_Float16)(d * rs * g[c] + bta[c]);
}

// ---- pack activations into WMMA A-fragment order (A 16x32: lane=16*half+M, e->K=16*half+e) ----
__global__ void pack_a_kernel(const _Float16* __restrict__ src, _Float16* __restrict__ dst,
                              int M, int Ksrc, int Kpad) {
  int idx = blockIdx.x * blockDim.x + threadIdx.x;
  if (idx >= M * Kpad) return;
  int KB = Kpad >> 5;
  int blk = idx >> 9, wi = idx & 511;
  int lane = wi >> 4, e = wi & 15;
  int mb = blk / KB, kb = blk % KB;
  int kloc = ((lane >> 4) << 4) + e;
  int mm = (mb << 4) + (lane & 15);
  int kk = (kb << 5) + kloc;
  dst[idx] = (kk < Ksrc) ? src[(size_t)mm * Ksrc + kk] : (_Float16)0.f;
}

// ---- pack weights W[n][k] into WMMA B-fragment order (B 32x16: lane=16*half+N, e->K) ----
__global__ void pack_b_kernel(const float* __restrict__ w, _Float16* __restrict__ dst,
                              int Nsrc, int Ksrc, int Npad, int Kpad) {
  int idx = blockIdx.x * blockDim.x + threadIdx.x;
  if (idx >= Npad * Kpad) return;
  int KB = Kpad >> 5;
  int blk = idx >> 9, wi = idx & 511;
  int lane = wi >> 4, e = wi & 15;
  int nb = blk / KB, kb = blk % KB;
  int nn = (nb << 4) + (lane & 15);
  int kk = (kb << 5) + ((lane >> 4) << 4) + e;
  float v = (nn < Nsrc && kk < Ksrc) ? w[(size_t)nn * Ksrc + kk] : 0.f;
  dst[idx] = (_Float16)v;
}

// ---- WMMA GEMM, one wave per 32x32 output tile (2x2 register blocking) ----
// mode 0: out f16 = acc+bias          (qkv)
// mode 1: out f16 = gelu(acc+bias)    (fc1)
// mode 2: x1[tok] = x[tok] + acc+bias (proj, window-reverse scatter)
// mode 3: out f32 = x1[m] + acc+bias  (fc2, token-major)
__global__ void wmma_gemm_kernel(const _Float16* __restrict__ Ap,
                                 const _Float16* __restrict__ Bp,
                                 const float* __restrict__ bias,
                                 int Mt2, int Nt2, int KB, int Nsrc,
                                 int mode, _Float16* __restrict__ outH, int ldH,
                                 const float* __restrict__ resid,
                                 float* __restrict__ outF) {
  int wid  = (int)((blockIdx.x * blockDim.x + threadIdx.x) >> 5);
  int lane = threadIdx.x & 31;
  if (wid >= Mt2 * Nt2) return;
  int mb2 = wid / Nt2, nb2 = wid % Nt2;
  const _Float16* a0p = Ap + (((size_t)(2 * mb2)     * KB * 32) + lane) * 16;
  const _Float16* a1p = Ap + (((size_t)(2 * mb2 + 1) * KB * 32) + lane) * 16;
  const _Float16* b0p = Bp + (((size_t)(2 * nb2)     * KB * 32) + lane) * 16;
  const _Float16* b1p = Bp + (((size_t)(2 * nb2 + 1) * KB * 32) + lane) * 16;
  v8f c00 = {}, c01 = {}, c10 = {}, c11 = {};
  for (int kb = 0; kb < KB; ++kb) {
    v16h a0 = *(const v16h*)a0p;  a0p += 512;
    v16h a1 = *(const v16h*)a1p;  a1p += 512;
    v16h b0 = *(const v16h*)b0p;  b0p += 512;
    v16h b1 = *(const v16h*)b1p;  b1p += 512;
    if (kb + 1 < KB) { __builtin_prefetch(a0p, 0, 2); __builtin_prefetch(a1p, 0, 2);
                       __builtin_prefetch(b0p, 0, 2); __builtin_prefetch(b1p, 0, 2); }
    c00 = wmma_f16(a0, b0, c00);
    c01 = wmma_f16(a0, b1, c01);
    c10 = wmma_f16(a1, b0, c10);
    c11 = wmma_f16(a1, b1, c11);
  }
  int half = lane >> 4, nloc = lane & 15;
  v8f cc[2][2] = { { c00, c01 }, { c10, c11 } };
#pragma unroll
  for (int i = 0; i < 2; ++i) {
#pragma unroll
    for (int j = 0; j < 2; ++j) {
      int n = ((2 * nb2 + j) << 4) + nloc;
      if (n >= Nsrc) continue;
      float bv = bias[n];
#pragma unroll
      for (int r = 0; r < 8; ++r) {
        int m = ((2 * mb2 + i) << 4) + r + (half << 3);
        float val = cc[i][j][r] + bv;
        if (mode == 0) {
          outH[(size_t)m * ldH + n] = (_Float16)val;
        } else if (mode == 1) {
          float gv = 0.5f * val * (1.f + erff(val * 0.70710678118654752f));
          outH[(size_t)m * ldH + n] = (_Float16)gv;
        } else if (mode == 2) {
          size_t tok = win_row_to_token(m);
          outF[tok * Cn + n] = resid[tok * Cn + n] + val;
        } else {
          outF[(size_t)m * Cn + n] = resid[(size_t)m * Cn + n] + val;
        }
      }
    }
  }
}

// ---- attention: one block (4 waves) per (window, head); WMMA for QK^T and PV ----
__global__ void attn_kernel(const _Float16* __restrict__ qkv,
                            const float* __restrict__ mask,
                            _Float16* __restrict__ aout) {
  __shared__ _Float16 qs[NTOK][32];
  __shared__ _Float16 ks[NTOK][32];
  __shared__ _Float16 vs[NTOK][32];
  __shared__ float    sS[4][16][NTOK];

  int win = blockIdx.x / HEADS;
  int hh  = blockIdx.x % HEADS;
  int tid = threadIdx.x;
  const float scale = 0.22360679775f;   // 20^-0.5

  for (int i = tid; i < NTOK * 32; i += blockDim.x) {
    int row = i >> 5, kk = i & 31;
    _Float16 qv = (_Float16)0.f, kv = qv, vv = qv;
    if (kk < HD) {
      size_t base = ((size_t)win * NTOK + row) * (3 * Cn);
      int col = hh * HD + kk;
      qv = (_Float16)((float)qkv[base + col] * scale);
      kv = qkv[base + Cn + col];
      vv = qkv[base + 2 * Cn + col];
    }
    qs[row][kk] = qv; ks[row][kk] = kv; vs[row][kk] = vv;
  }
  __syncthreads();

  int wave = tid >> 5, lane = tid & 31;
  int half = lane >> 4, nloc = lane & 15, mloc = lane & 15;
  int maskw = win % NW_PER_B;

  for (int it = 0; it < 2; ++it) {
    int rb = wave + 4 * it;
    int row0 = rb << 4;
    // A fragment: q rows (K padded to 32)
    v16h aq;
#pragma unroll
    for (int e = 0; e < 16; ++e) {
      int kloc = (half << 4) + e;
      aq[e] = qs[row0 + mloc][kloc];
    }
    // S = q kT  (8 column tiles)
    for (int nb = 0; nb < 8; ++nb) {
      v16h bk;
#pragma unroll
      for (int e = 0; e < 16; ++e) bk[e] = ks[(nb << 4) + nloc][(half << 4) + e];
      v8f s = {};
      s = wmma_f16(aq, bk, s);
#pragma unroll
      for (int r = 0; r < 8; ++r) sS[wave][r + (half << 3)][(nb << 4) + nloc] = s[r];
    }
    // mask add + softmax (per row, wave32 reductions)
    for (int row = 0; row < 16; ++row) {
      const float* mrow = mask + ((size_t)maskw * NTOK + (row0 + row)) * NTOK;
      float v[4]; float mx = -3.0e38f;
#pragma unroll
      for (int j = 0; j < 4; ++j) {
        int col = lane + (j << 5);
        float t = sS[wave][row][col] + mrow[col];
        v[j] = t; mx = fmaxf(mx, t);
      }
#pragma unroll
      for (int off = 16; off > 0; off >>= 1) mx = fmaxf(mx, __shfl_xor(mx, off, 32));
      float sum = 0.f;
#pragma unroll
      for (int j = 0; j < 4; ++j) { v[j] = __expf(v[j] - mx); sum += v[j]; }
#pragma unroll
      for (int off = 16; off > 0; off >>= 1) sum += __shfl_xor(sum, off, 32);
      float inv = 1.f / sum;
#pragma unroll
      for (int j = 0; j < 4; ++j) sS[wave][row][lane + (j << 5)] = v[j] * inv;
    }
    // O = P V  (K=128 -> 4 steps; 2 column tiles cover hd=20)
    for (int nb2 = 0; nb2 < 2; ++nb2) {
      v8f o = {};
      for (int kb = 0; kb < 4; ++kb) {
        v16h ap;
#pragma unroll
        for (int e = 0; e < 16; ++e) {
          int kloc = (half << 4) + e;
          ap[e] = (_Float16)sS[wave][mloc][(kb << 5) + kloc];
        }
        v16h bv;
#pragma unroll
        for (int e = 0; e < 16; ++e) bv[e] = vs[(kb << 5) + (half << 4) + e][(nb2 << 4) + nloc];
        o = wmma_f16(ap, bv, o);
      }
      int d = (nb2 << 4) + nloc;
      if (d < HD) {
#pragma unroll
        for (int r = 0; r < 8; ++r) {
          int mr = row0 + r + (half << 3);
          aout[((size_t)win * NTOK + mr) * Cn + hh * HD + d] = (_Float16)o[r];
        }
      }
    }
  }
}

extern "C" void kernel_launch(void* const* d_in, const int* in_sizes, int n_in,
                              void* d_out, int out_size, void* d_ws, size_t ws_size,
                              hipStream_t stream) {
  (void)in_sizes; (void)n_in; (void)out_size; (void)ws_size;
  const float* x      = (const float*)d_in[0];
  const float* g1     = (const float*)d_in[1];
  const float* b1     = (const float*)d_in[2];
  const float* qkv_w  = (const float*)d_in[3];
  const float* qkv_b  = (const float*)d_in[4];
  const float* proj_w = (const float*)d_in[5];
  const float* proj_b = (const float*)d_in[6];
  const float* g2     = (const float*)d_in[7];
  const float* b2     = (const float*)d_in[8];
  const float* fc1_w  = (const float*)d_in[9];
  const float* fc1_b  = (const float*)d_in[10];
  const float* fc2_w  = (const float*)d_in[11];
  const float* fc2_b  = (const float*)d_in[12];
  const float* amask  = (const float*)d_in[13];
  float* out = (float*)d_out;

  char* ws = (char*)d_ws;
  size_t off = 0;
  auto take = [&](size_t bytes) -> void* {
    void* p = ws + off; off = (off + bytes + 255) & ~(size_t)255; return p;
  };
  _Float16* packA = (_Float16*)take((size_t)MTOT * 256 * 2);   // A fragments (max K=256)
  _Float16* wqP   = (_Float16*)take((size_t)384 * 128 * 2);    // qkv_w packed (Npad 384)
  _Float16* wpP   = (_Float16*)take((size_t)128 * 128 * 2);    // proj_w packed
  _Float16* w1P   = (_Float16*)take((size_t)256 * 128 * 2);    // fc1_w packed (Npad 256)
  _Float16* w2P   = (_Float16*)take((size_t)128 * 256 * 2);    // fc2_w packed
  _Float16* qkvB  = (_Float16*)take((size_t)MTOT * 360 * 2);   // qkv activations f16
  _Float16* act1  = (_Float16*)take((size_t)MTOT * HID * 2);   // fc1 output
  _Float16* act2  = (_Float16*)take((size_t)MTOT * Cn * 2);    // xln1 / attn_out / xln2
  float*    x1    = (float*)take((size_t)MTOT * Cn * 4);       // residual state

  auto blocks = [](long long total, int thr) { return dim3((unsigned)((total + thr - 1) / thr)); };
  const int Mt2 = MTOT / 32;   // 3072 row-pair tiles

  // weight packing (tiny)
  pack_b_kernel<<<blocks(384 * 128, 256), 256, 0, stream>>>(qkv_w, wqP, 360, 120, 384, 128);
  pack_b_kernel<<<blocks(128 * 128, 256), 256, 0, stream>>>(proj_w, wpP, 120, 120, 128, 128);
  pack_b_kernel<<<blocks(256 * 128, 256), 256, 0, stream>>>(fc1_w, w1P, 240, 120, 256, 128);
  pack_b_kernel<<<blocks(128 * 256, 256), 256, 0, stream>>>(fc2_w, w2P, 120, 240, 128, 256);

  // LN1 fused with roll(-shift)+window partition (window-order rows)
  ln_kernel<<<MTOT, 128, 0, stream>>>(x, g1, b1, act2, 1);
  pack_a_kernel<<<blocks((long long)MTOT * 128, 256), 256, 0, stream>>>(act2, packA, MTOT, 120, 128);

  // qkv = xw @ qkv_w.T + b  -> f16 [M,360]
  { long long waves = (long long)Mt2 * 12;
    wmma_gemm_kernel<<<blocks(waves * 32, 256), 256, 0, stream>>>(
        packA, wqP, qkv_b, Mt2, 12, 4, 360, 0, qkvB, 360, nullptr, nullptr); }

  // windowed multi-head attention
  attn_kernel<<<NWIN * HEADS, 128, 0, stream>>>(qkvB, amask, act2);
  pack_a_kernel<<<blocks((long long)MTOT * 128, 256), 256, 0, stream>>>(act2, packA, MTOT, 120, 128);

  // proj + window reverse + unshift + residual -> x1 (f32, token order)
  { long long waves = (long long)Mt2 * 4;
    wmma_gemm_kernel<<<blocks(waves * 32, 256), 256, 0, stream>>>(
        packA, wpP, proj_b, Mt2, 4, 4, 120, 2, nullptr, 0, x, x1); }

  // LN2 (token order) -> MLP
  ln_kernel<<<MTOT, 128, 0, stream>>>(x1, g2, b2, act2, 0);
  pack_a_kernel<<<blocks((long long)MTOT * 128, 256), 256, 0, stream>>>(act2, packA, MTOT, 120, 128);

  { long long waves = (long long)Mt2 * 8;   // fc1 + GELU
    wmma_gemm_kernel<<<blocks(waves * 32, 256), 256, 0, stream>>>(
        packA, w1P, fc1_b, Mt2, 8, 4, 240, 1, act1, 240, nullptr, nullptr); }

  pack_a_kernel<<<blocks((long long)MTOT * 256, 256), 256, 0, stream>>>(act1, packA, MTOT, 240, 256);

  { long long waves = (long long)Mt2 * 4;   // fc2 + residual -> out f32
    wmma_gemm_kernel<<<blocks(waves * 32, 256), 256, 0, stream>>>(
        packA, w2P, fc2_b, Mt2, 4, 8, 120, 3, nullptr, 0, x1, out); }
}
